// EntityPredictor_11510512353536
// MI455X (gfx1250) — matrix operand order repaired and checked by reading, not compiled, not run on hardware
//
#include <hip/hip_runtime.h>
#include <hip/hip_bf16.h>
#include <math.h>

// ---------------------------------------------------------------------------
// EntityPredictor forward on gfx1250 (MI455X), wave32 + WMMA f16->f32.
// B=64, P=32, E=8192, H=1024, NH=4, DH=256.
// ---------------------------------------------------------------------------

#define BB 64
#define PP 32
#define EE 8192
#define HH 1024
#define NHH 4
#define DHH 256
#define MM (BB * PP)            // 2048 rows for token-level GEMMs
#define EPSF 1e-5f
#define ATT_SCALE 0.0625f       // 1/sqrt(256)

typedef _Float16 h8   __attribute__((ext_vector_type(8)));
typedef _Float16 v16h __attribute__((ext_vector_type(16)));
typedef float    v8f  __attribute__((ext_vector_type(8)));

// ---------------------------------------------------------------------------
// Fragment loaders.
// A: f16 row-major [M][K].  Lane l (0..31) holds row (l&15).
//   VGPRs 0-3: K = klo+0..7, VGPRs 4-7: K = 16+klo+0..7, klo = (l<16)?0:8.
// B: staged as B^T, f16 row-major [N][K].  Lane l holds column (l&15).
//   Same K order as A: VGPRs 0-3: K = klo+0..7, VGPRs 4-7: K = 16+klo+0..7.
// ---------------------------------------------------------------------------
__device__ __forceinline__ v16h load_a_frag(const _Float16* A, int ldk,
                                            int row, int k0, int lane) {
    const _Float16* p = A + (size_t)row * ldk + k0 + ((lane >> 4) << 3);
    h8 lo = *(const h8*)(p);
    h8 hi = *(const h8*)(p + 16);
    return __builtin_shufflevector(lo, hi, 0, 1, 2, 3, 4, 5, 6, 7,
                                   8, 9, 10, 11, 12, 13, 14, 15);
}

__device__ __forceinline__ v16h load_b_frag(const _Float16* BT, int ldk,
                                            int col, int k0, int lane) {
    const _Float16* p = BT + (size_t)col * ldk + k0 + ((lane >> 4) << 3);
    h8 lo = *(const h8*)(p);
    h8 hi = *(const h8*)(p + 16);
    return __builtin_shufflevector(lo, hi, 0, 1, 2, 3, 4, 5, 6, 7,
                                   8, 9, 10, 11, 12, 13, 14, 15);
}

// ---------------------------------------------------------------------------
// WMMA GEMM: C[M][N] = A[M][K] * BT[N][K]^T  (+ epilogue)
//   MODE 0: Ch[m][n] = (f16)(acc * invcnt[m])          (pooling)
//   MODE 1: Cf[m][n] = acc + bias[n]                   (qkv / proj)
// Block = (32,8): 8 waves, each wave computes a 16x64 tile; block = 128x64.
// Grid = (N/64, M/128).  All dims divide exactly for our sizes.
// ---------------------------------------------------------------------------
template <int MODE>
__global__ __launch_bounds__(256)
void wmma_gemm_kernel(const _Float16* __restrict__ A,
                      const _Float16* __restrict__ BT,
                      float* __restrict__ Cf, _Float16* __restrict__ Ch,
                      const float* __restrict__ bias,
                      const float* __restrict__ invcnt,
                      int M, int N, int K) {
    const int lane = threadIdx.x;                 // 0..31
    const int wave = threadIdx.y;                 // 0..7
    const int m0 = blockIdx.y * 128 + wave * 16;
    const int n0 = blockIdx.x * 64;
    const int row = m0 + (lane & 15);
    const int colA = n0 + (lane & 15);

    v8f acc0 = {}, acc1 = {}, acc2 = {}, acc3 = {};

    for (int k = 0; k < K; k += 32) {
        v16h a  = load_a_frag(A, K, row, k, lane);
        v16h b0 = load_b_frag(BT, K, colA,      k, lane);
        v16h b1 = load_b_frag(BT, K, colA + 16, k, lane);
        v16h b2 = load_b_frag(BT, K, colA + 32, k, lane);
        v16h b3 = load_b_frag(BT, K, colA + 48, k, lane);
        acc0 = __builtin_amdgcn_wmma_f32_16x16x32_f16(false, a, false, b0,
                                                      (short)0, acc0, false, false);
        acc1 = __builtin_amdgcn_wmma_f32_16x16x32_f16(false, a, false, b1,
                                                      (short)0, acc1, false, false);
        acc2 = __builtin_amdgcn_wmma_f32_16x16x32_f16(false, a, false, b2,
                                                      (short)0, acc2, false, false);
        acc3 = __builtin_amdgcn_wmma_f32_16x16x32_f16(false, a, false, b3,
                                                      (short)0, acc3, false, false);
    }

    // C/D layout: VGPR v -> row m0 + v + 8*(lane>=16), col n0 + (lane&15) + 16*t
    const int rbase = m0 + ((lane >> 4) << 3);
    const int cbase = n0 + (lane & 15);
#pragma unroll
    for (int v = 0; v < 8; ++v) {
        const int r = rbase + v;
        if (MODE == 0) {
            const float s = invcnt[r];
            _Float16* p = Ch + (size_t)r * N + cbase;
            p[0]  = (_Float16)(acc0[v] * s);
            p[16] = (_Float16)(acc1[v] * s);
            p[32] = (_Float16)(acc2[v] * s);
            p[48] = (_Float16)(acc3[v] * s);
        } else {
            float* p = Cf + (size_t)r * N + cbase;
            p[0]  = acc0[v] + bias[cbase];
            p[16] = acc1[v] + bias[cbase + 16];
            p[32] = acc2[v] + bias[cbase + 32];
            p[48] = acc3[v] + bias[cbase + 48];
        }
    }
}

// ---------------------------------------------------------------------------
// mask = (inputs == 1) as f16, plus invcnt[row] = 1/max(sum(mask),1)
// One block of 256 threads per row (row = b*P+p, 2048 rows).
// ---------------------------------------------------------------------------
__global__ __launch_bounds__(256)
void mask_cnt_kernel(const int* __restrict__ in, _Float16* __restrict__ maskh,
                     float* __restrict__ invcnt) {
    const int row = blockIdx.x;
    const int* rp = in + (size_t)row * EE;
    _Float16* mp = maskh + (size_t)row * EE;
    int cnt = 0;
    for (int e = threadIdx.x; e < EE; e += 256) {
        const int v = (rp[e] == 1) ? 1 : 0;
        mp[e] = (_Float16)v;
        cnt += v;
    }
    __shared__ int red[256];
    red[threadIdx.x] = cnt;
    __syncthreads();
    for (int s = 128; s > 0; s >>= 1) {
        if (threadIdx.x < s) red[threadIdx.x] += red[threadIdx.x + s];
        __syncthreads();
    }
    if (threadIdx.x == 0) {
        const float c = (float)red[0];
        invcnt[row] = 1.0f / fmaxf(c, 1.0f);
    }
}

// ---------------------------------------------------------------------------
// Tiled transpose + f32->f16: dst[C][R] = (f16) src[R][C].  Block (32,8).
// ---------------------------------------------------------------------------
__global__ __launch_bounds__(256)
void transpose_f32_f16_kernel(const float* __restrict__ src,
                              _Float16* __restrict__ dst, int R, int C) {
    __shared__ float tile[32][33];
    const int c0 = blockIdx.x * 32;
    const int r0 = blockIdx.y * 32;
    for (int i = threadIdx.y; i < 32; i += 8)
        tile[i][threadIdx.x] = src[(size_t)(r0 + i) * C + c0 + threadIdx.x];
    __syncthreads();
    for (int i = threadIdx.y; i < 32; i += 8)
        dst[(size_t)(c0 + i) * R + r0 + threadIdx.x] =
            (_Float16)tile[threadIdx.x][i];
}

// f32 -> f16 elementwise copy (weights already stored [N][K] = BT layout)
__global__ __launch_bounds__(256)
void convert_f16_kernel(const float* __restrict__ src,
                        _Float16* __restrict__ dst, int n) {
    const int i = blockIdx.x * 256 + threadIdx.x;
    if (i < n) dst[i] = (_Float16)src[i];
}

// ---------------------------------------------------------------------------
// Attention: one thread per (b, head, query i).  P=32 scores live in regs.
// qkv layout [2048][3072]: q | k | v, each H=1024 = 4 heads * 256.
// Output oh f16 [2048][1024] (input A of the wout GEMM).
// ---------------------------------------------------------------------------
__global__ __launch_bounds__(256)
void attention_kernel(const float* __restrict__ qkv, _Float16* __restrict__ oh) {
    const int t = blockIdx.x * 256 + threadIdx.x;   // 0..8191
    const int b = t >> 7;
    const int r = t & 127;
    const int h = r >> 5;
    const int i = r & 31;

    const float* q  = qkv + (size_t)(b * PP + i) * (3 * HH) + h * DHH;
    const float* kb = qkv + (size_t)(b * PP) * (3 * HH) + HH + h * DHH;
    const float* vb = qkv + (size_t)(b * PP) * (3 * HH) + 2 * HH + h * DHH;

    float sc[PP];
    float mx = -3.0e38f;
#pragma unroll 1
    for (int j = 0; j < PP; ++j) {
        const float* kp = kb + (size_t)j * (3 * HH);
        float s = 0.0f;
        for (int d = 0; d < DHH; ++d) s += q[d] * kp[d];
        s *= ATT_SCALE;
        sc[j] = s;
        mx = fmaxf(mx, s);
    }
    float denom = 0.0f;
    for (int j = 0; j < PP; ++j) {
        sc[j] = expf(sc[j] - mx);
        denom += sc[j];
    }
    const float inv = 1.0f / denom;
    _Float16* op = oh + (size_t)(b * PP + i) * HH + h * DHH;
#pragma unroll 1
    for (int d = 0; d < DHH; ++d) {
        float acc = 0.0f;
        for (int j = 0; j < PP; ++j) acc += sc[j] * vb[(size_t)j * (3 * HH) + d];
        op[d] = (_Float16)(acc * inv);
    }
}

// ---------------------------------------------------------------------------
// Mean over P then LayerNorm(ln1): xln[b][c], one block per b.
// ---------------------------------------------------------------------------
__global__ __launch_bounds__(256)
void mean_ln_kernel(const float* __restrict__ proj, const float* __restrict__ g,
                    const float* __restrict__ be, float* __restrict__ xln) {
    const int b = blockIdx.x;
    __shared__ float xs[HH];
    __shared__ float red[256];
    for (int c = threadIdx.x; c < HH; c += 256) {
        float s = 0.0f;
        for (int p = 0; p < PP; ++p) s += proj[(size_t)(b * PP + p) * HH + c];
        xs[c] = s * (1.0f / (float)PP);
    }
    __syncthreads();
    float ls = 0.0f;
    for (int c = threadIdx.x; c < HH; c += 256) ls += xs[c];
    red[threadIdx.x] = ls;
    __syncthreads();
    for (int s = 128; s > 0; s >>= 1) {
        if (threadIdx.x < s) red[threadIdx.x] += red[threadIdx.x + s];
        __syncthreads();
    }
    const float mean = red[0] * (1.0f / (float)HH);
    __syncthreads();
    float lv = 0.0f;
    for (int c = threadIdx.x; c < HH; c += 256) {
        const float d = xs[c] - mean;
        lv += d * d;
    }
    red[threadIdx.x] = lv;
    __syncthreads();
    for (int s = 128; s > 0; s >>= 1) {
        if (threadIdx.x < s) red[threadIdx.x] += red[threadIdx.x + s];
        __syncthreads();
    }
    const float rs = rsqrtf(red[0] * (1.0f / (float)HH) + EPSF);
    for (int c = threadIdx.x; c < HH; c += 256)
        xln[(size_t)b * HH + c] = (xs[c] - mean) * rs * g[c] + be[c];
}

// ---------------------------------------------------------------------------
// LayerNorm(ln2) then BatchNorm-eval(bn1), in place on h[64][D].
// ---------------------------------------------------------------------------
__global__ __launch_bounds__(256)
void ln_bn_kernel(float* __restrict__ hrow, const float* __restrict__ lg,
                  const float* __restrict__ lb, const float* __restrict__ bg,
                  const float* __restrict__ bb, int D) {
    const int b = blockIdx.x;
    float* row = hrow + (size_t)b * D;
    __shared__ float red[256];
    float ls = 0.0f;
    for (int c = threadIdx.x; c < D; c += 256) ls += row[c];
    red[threadIdx.x] = ls;
    __syncthreads();
    for (int s = 128; s > 0; s >>= 1) {
        if (threadIdx.x < s) red[threadIdx.x] += red[threadIdx.x + s];
        __syncthreads();
    }
    const float mean = red[0] / (float)D;
    __syncthreads();
    float lv = 0.0f;
    for (int c = threadIdx.x; c < D; c += 256) {
        const float d = row[c] - mean;
        lv += d * d;
    }
    red[threadIdx.x] = lv;
    __syncthreads();
    for (int s = 128; s > 0; s >>= 1) {
        if (threadIdx.x < s) red[threadIdx.x] += red[threadIdx.x + s];
        __syncthreads();
    }
    const float rs = rsqrtf(red[0] / (float)D + EPSF);
    const float bnrs = rsqrtf(1.0f + EPSF);
    for (int c = threadIdx.x; c < D; c += 256) {
        const float ln = (row[c] - mean) * rs * lg[c] + lb[c];
        row[c] = ln * bnrs * bg[c] + bb[c];
    }
}

// ---------------------------------------------------------------------------
// Small fully-connected: out[b][j] = act(dot(in[b], W[j]) + bias[j])
//   mode 1: relu;  mode 2: relu then bn2-eval;  mode 3: sigmoid
// One thread per output element (M=64 rows only; memory bound on W).
// ---------------------------------------------------------------------------
__global__ __launch_bounds__(256)
void fc_kernel(const float* __restrict__ in, const float* __restrict__ W,
               const float* __restrict__ bias, float* __restrict__ out,
               int Bn, int J, int Kd, int mode,
               const float* __restrict__ bng, const float* __restrict__ bnb) {
    const int idx = blockIdx.x * 256 + threadIdx.x;
    if (idx >= Bn * J) return;
    const int b = idx / J;
    const int j = idx - b * J;
    const float* ip = in + (size_t)b * Kd;
    const float* wp = W + (size_t)j * Kd;
    float s = bias[j];
    for (int k = 0; k < Kd; ++k) s += ip[k] * wp[k];
    if (mode == 1) {
        s = fmaxf(s, 0.0f);
    } else if (mode == 2) {
        s = fmaxf(s, 0.0f);
        s = s * rsqrtf(1.0f + EPSF) * bng[j] + bnb[j];
    } else if (mode == 3) {
        s = 1.0f / (1.0f + expf(-s));
    }
    out[idx] = s;
}

// ---------------------------------------------------------------------------
// Host-side launch orchestration.
// ---------------------------------------------------------------------------
extern "C" void kernel_launch(void* const* d_in, const int* in_sizes, int n_in,
                              void* d_out, int out_size, void* d_ws, size_t ws_size,
                              hipStream_t stream) {
    const int*   inputs = (const int*)d_in[0];
    const float* emb    = (const float*)d_in[1];
    const float* win    = (const float*)d_in[2];
    const float* bin_   = (const float*)d_in[3];
    const float* wout   = (const float*)d_in[4];
    const float* bout   = (const float*)d_in[5];
    const float* fc1_w  = (const float*)d_in[6];
    const float* fc1_b  = (const float*)d_in[7];
    const float* fc2_w  = (const float*)d_in[8];
    const float* fc2_b  = (const float*)d_in[9];
    const float* fco_w  = (const float*)d_in[10];
    const float* fco_b  = (const float*)d_in[11];
    const float* ln1_g  = (const float*)d_in[12];
    const float* ln1_b  = (const float*)d_in[13];
    const float* ln2_g  = (const float*)d_in[14];
    const float* ln2_b  = (const float*)d_in[15];
    const float* bn1_g  = (const float*)d_in[16];
    const float* bn1_b  = (const float*)d_in[17];
    const float* bn2_g  = (const float*)d_in[18];
    const float* bn2_b  = (const float*)d_in[19];
    float* out = (float*)d_out;

    // Workspace carve-up (all offsets 256B-aligned).
    char* ws = (char*)d_ws;
    _Float16* maskh   = (_Float16*)(ws + 0);                     // 2048*8192 f16
    _Float16* embT    = (_Float16*)(ws + 33554432);              // 1024*8192 f16
    float*    invcnt  = (float*)   (ws + 50331648);              // 2048 f32
    _Float16* pooledh = (_Float16*)(ws + 50339840);              // 2048*1024 f16
    _Float16* winh    = (_Float16*)(ws + 54534144);              // 3072*1024 f16
    float*    qkv     = (float*)   (ws + 60825600);              // 2048*3072 f32
    _Float16* oh      = (_Float16*)(ws + 85991424);              // 2048*1024 f16
    _Float16* wouth   = (_Float16*)(ws + 90185728);              // 1024*1024 f16
    float*    proj    = (float*)   (ws + 92282880);              // 2048*1024 f32
    float*    xln     = (float*)   (ws + 100671488);             // 64*1024 f32
    float*    h1      = (float*)   (ws + 100933632);             // 64*2048 f32
    float*    h2      = (float*)   (ws + 101457920);             // 64*1024 f32
    (void)ws_size; (void)in_sizes; (void)n_in; (void)out_size;

    // 1. mask (f16) + 1/max(cnt,1)
    mask_cnt_kernel<<<MM, 256, 0, stream>>>(inputs, maskh, invcnt);

    // 2. emb [E][H] f32 -> embT [H][E] f16  (BT layout for GEMM1)
    transpose_f32_f16_kernel<<<dim3(HH / 32, EE / 32), dim3(32, 8), 0, stream>>>(
        emb, embT, EE, HH);

    // 3. weight f32 -> f16 (already [N][K] layout)
    convert_f16_kernel<<<(3 * HH * HH + 255) / 256, 256, 0, stream>>>(
        win, winh, 3 * HH * HH);
    convert_f16_kernel<<<(HH * HH + 255) / 256, 256, 0, stream>>>(
        wout, wouth, HH * HH);

    // 4. pooled(f16) = (mask @ emb) * invcnt   [2048 x 1024], K=8192
    wmma_gemm_kernel<0><<<dim3(HH / 64, MM / 128), dim3(32, 8), 0, stream>>>(
        maskh, embT, nullptr, pooledh, nullptr, invcnt, MM, HH, EE);

    // 5. qkv(f32) = pooled @ win^T + bin       [2048 x 3072], K=1024
    wmma_gemm_kernel<1><<<dim3(3 * HH / 64, MM / 128), dim3(32, 8), 0, stream>>>(
        pooledh, winh, qkv, nullptr, bin_, nullptr, MM, 3 * HH, HH);

    // 6. attention -> oh (f16)                 [2048 x 1024]
    attention_kernel<<<(BB * NHH * PP) / 256, 256, 0, stream>>>(qkv, oh);

    // 7. proj(f32) = o @ wout^T + bout         [2048 x 1024], K=1024
    wmma_gemm_kernel<1><<<dim3(HH / 64, MM / 128), dim3(32, 8), 0, stream>>>(
        oh, wouth, proj, nullptr, bout, nullptr, MM, HH, HH);

    // 8. mean over P + LayerNorm(ln1) -> xln [64 x 1024]
    mean_ln_kernel<<<BB, 256, 0, stream>>>(proj, ln1_g, ln1_b, xln);

    // 9. h1 = relu(xln @ fc1_w^T + fc1_b)      [64 x 2048]
    fc_kernel<<<(BB * 2 * HH + 255) / 256, 256, 0, stream>>>(
        xln, fc1_w, fc1_b, h1, BB, 2 * HH, HH, 1, nullptr, nullptr);

    // 10. h1 = bn1(ln2(h1)) in place
    ln_bn_kernel<<<BB, 256, 0, stream>>>(h1, ln2_g, ln2_b, bn1_g, bn1_b, 2 * HH);

    // 11. h2 = bn2(relu(h1 @ fc2_w^T + fc2_b)) [64 x 1024]
    fc_kernel<<<(BB * HH + 255) / 256, 256, 0, stream>>>(
        h1, fc2_w, fc2_b, h2, BB, HH, 2 * HH, 2, bn2_g, bn2_b);

    // 12. out = sigmoid(h2 @ fco_w^T + fco_b)  [64 x 8192]
    fc_kernel<<<(BB * EE + 255) / 256, 256, 0, stream>>>(
        h2, fco_w, fco_b, out, BB, EE, HH, 3, nullptr, nullptr);
}
